// VQEmbedding_57243324121649
// MI455X (gfx1250) — compile-verified, matrix operand order checked by reading, not run on hardware
//
#include <hip/hip_runtime.h>
#include <hip/hip_bf16.h>

// ---------------------------------------------------------------------------
// VQ codebook lookup, fused distance-GEMM + argmin, for gfx1250 (MI455X).
//   z_e_x : [4, 256, 32, 64] f32   (d_in[0])
//   codebook : [8192, 256] f32     (d_in[1])
// Outputs (concatenated, float):
//   indices  [8192]           (as float)
//   z_q_x_bar[4,256,32,64]    (2097152 floats)
//   commitment_loss [1]
// ---------------------------------------------------------------------------

typedef __attribute__((ext_vector_type(16))) __bf16          v16bf;
typedef __attribute__((ext_vector_type(8)))  float           v8f;
typedef __attribute__((ext_vector_type(16))) unsigned short  us16;
typedef __attribute__((ext_vector_type(8)))  unsigned short  us8;

#define VQ_D     256
#define VQ_K     8192
#define VQ_N     8192
#define VQ_NT    (VQ_K / 16)     // 512 column tiles
#define PLANE    2048            // 32*64, spatial plane of z_e
#define BSTRIDE  524288          // 256*2048, batch stride of z_e

__device__ __forceinline__ void split_bf16(float f, unsigned short& hi, unsigned short& lo) {
  unsigned u = __float_as_uint(f);
  hi = (unsigned short)(u >> 16);                       // truncated bf16 high part
  float hif = __uint_as_float(u & 0xffff0000u);
  lo = (unsigned short)(__float_as_uint(f - hif) >> 16); // bf16 of the residual
}

// --- Kernel 1: c_sqr[k] = ||codebook[k]||^2 (one wave per code row) ---------
__global__ void vq_csqr(const float* __restrict__ cb, float* __restrict__ csqr) {
  int wv   = (blockIdx.x * blockDim.x + threadIdx.x) >> 5;
  int lane = threadIdx.x & 31;
  const float4* r4 = (const float4*)(cb + (size_t)wv * VQ_D + lane * 8);
  float4 a = r4[0], b = r4[1];
  float s = a.x*a.x + a.y*a.y + a.z*a.z + a.w*a.w
          + b.x*b.x + b.y*b.y + b.z*b.z + b.w*b.w;
  #pragma unroll
  for (int m = 16; m >= 1; m >>= 1) s += __shfl_xor(s, m, 32);
  if (lane == 0) csqr[wv] = s;
}

// --- Kernel 2: fused x.c^T GEMM (split-bf16 WMMA) + running argmin ----------
// Wave owns 16 rows of x (A tile resident in VGPRs, hi+lo planes).
// Workgroup of 4 waves shares one 16-code B tile staged in LDS per iteration.
// Two independent WMMA accumulator chains (even/odd k-steps) for ILP.
__global__ __launch_bounds__(128) void vq_argmin(
    const float* __restrict__ ze, const float* __restrict__ cb,
    const float* __restrict__ csqr,
    float* __restrict__ out_idx_f, int* __restrict__ ws_idx) {
  __shared__ unsigned short sBhi[16 * VQ_D];   // 8 KB
  __shared__ unsigned short sBlo[16 * VQ_D];   // 8 KB

  const int tid   = threadIdx.x;
  const int lane  = tid & 31;
  const int wave  = tid >> 5;
  const int khalf = lane >> 4;     // which K-half this lane holds (A/B layout)
  const int lcol  = lane & 15;

  // A tile: row n, ISA 16-bit A layout: lanes0-15 K={e}+{16..23}, lanes16-31 +8
  const int  n    = blockIdx.x * 64 + wave * 16 + lcol;
  const long base = (long)(n >> 11) * BSTRIDE + (n & (PLANE - 1));

  us16 a_hi[8], a_lo[8];
  #pragma unroll
  for (int ks = 0; ks < 8; ++ks) {
    #pragma unroll
    for (int e = 0; e < 16; ++e) {
      int d = ks * 32 + khalf * 8 + (e < 8 ? e : e + 8);
      unsigned short hi, lo;
      split_bf16(ze[base + (long)d * PLANE], hi, lo);
      a_hi[ks][e] = hi; a_lo[ks][e] = lo;
    }
  }

  float best[8]; int bidx[8];
  #pragma unroll
  for (int v = 0; v < 8; ++v) { best[v] = 3.4e38f; bidx[v] = 0; }

  for (int nt = 0; nt < VQ_NT; ++nt) {
    __syncthreads();                      // previous tile's readers done
    {                                     // stage 16 codebook rows -> LDS
      int row = tid >> 3;                 // 0..15
      int db  = (tid & 7) * 32;           // 0..224
      const float* src = cb + (size_t)(nt * 16 + row) * VQ_D + db;
      #pragma unroll
      for (int j = 0; j < 32; ++j) {
        unsigned short hi, lo;
        split_bf16(src[j], hi, lo);
        sBhi[row * VQ_D + db + j] = hi;
        sBlo[row * VQ_D + db + j] = lo;
      }
    }
    __syncthreads();

    v8f acc0 = {0.f, 0.f, 0.f, 0.f, 0.f, 0.f, 0.f, 0.f};
    v8f acc1 = {0.f, 0.f, 0.f, 0.f, 0.f, 0.f, 0.f, 0.f};
    #pragma unroll
    for (int ks = 0; ks < 8; ++ks) {
      int off = lcol * VQ_D + ks * 32 + khalf * 8;
      us8 bh0 = *(const us8*)&sBhi[off];
      us8 bh1 = *(const us8*)&sBhi[off + 16];
      us8 bl0 = *(const us8*)&sBlo[off];
      us8 bl1 = *(const us8*)&sBlo[off + 16];
      v16bf bh = __builtin_bit_cast(v16bf, __builtin_shufflevector(
          bh0, bh1, 0,1,2,3,4,5,6,7,8,9,10,11,12,13,14,15));
      v16bf bl = __builtin_bit_cast(v16bf, __builtin_shufflevector(
          bl0, bl1, 0,1,2,3,4,5,6,7,8,9,10,11,12,13,14,15));
      v16bf ah = __builtin_bit_cast(v16bf, a_hi[ks]);
      v16bf al = __builtin_bit_cast(v16bf, a_lo[ks]);
      // split-bf16: x.c ~= lo*hi + hi*lo + hi*hi, fp32 accumulate.
      // Alternate between two accumulators to break the D->C RAW chain.
      if (ks & 1) {
        acc1 = __builtin_amdgcn_wmma_f32_16x16x32_bf16(false, al, false, bh, (short)0, acc1, false, false);
        acc1 = __builtin_amdgcn_wmma_f32_16x16x32_bf16(false, ah, false, bl, (short)0, acc1, false, false);
        acc1 = __builtin_amdgcn_wmma_f32_16x16x32_bf16(false, ah, false, bh, (short)0, acc1, false, false);
      } else {
        acc0 = __builtin_amdgcn_wmma_f32_16x16x32_bf16(false, al, false, bh, (short)0, acc0, false, false);
        acc0 = __builtin_amdgcn_wmma_f32_16x16x32_bf16(false, ah, false, bl, (short)0, acc0, false, false);
        acc0 = __builtin_amdgcn_wmma_f32_16x16x32_bf16(false, ah, false, bh, (short)0, acc0, false, false);
      }
    }

    // score = ||c||^2 - 2 x.c  (||x||^2 constant per row, irrelevant to argmin)
    const int   code = nt * 16 + lcol;
    const float cs   = csqr[code];
    #pragma unroll
    for (int v = 0; v < 8; ++v) {
      float dot = acc0[v] + acc1[v];
      float s = __builtin_fmaf(-2.f, dot, cs);
      if (s < best[v]) { best[v] = s; bidx[v] = code; }  // strict: keeps first
    }
  }

  // cross-lane argmin over the 16 column slots in each half-wave
  #pragma unroll
  for (int m = 1; m < 16; m <<= 1) {
    #pragma unroll
    for (int v = 0; v < 8; ++v) {
      float ob = __shfl_xor(best[v], m, 32);
      int   oi = __shfl_xor(bidx[v], m, 32);
      if (ob < best[v] || (ob == best[v] && oi < bidx[v])) { best[v] = ob; bidx[v] = oi; }
    }
  }
  if (lcol == 0) {                       // C layout: VGPR v -> row v + 8*khalf
    #pragma unroll
    for (int v = 0; v < 8; ++v) {
      int row = blockIdx.x * 64 + wave * 16 + v + khalf * 8;
      out_idx_f[row] = (float)bidx[v];
      ws_idx[row]    = bidx[v];
    }
  }
}

// --- Kernel 3: gather z_q into [b,D,h,t] layout + per-row squared-error -----
__global__ void vq_gather(const float* __restrict__ ze, const float* __restrict__ cb,
                          const int* __restrict__ ws_idx,
                          float* __restrict__ out_zq, float* __restrict__ blocksums) {
  __shared__ float partial[8];
  int n = blockIdx.x, d = threadIdx.x;
  int idx = ws_idx[n];
  float c = cb[(size_t)idx * VQ_D + d];
  long pos = (long)(n >> 11) * BSTRIDE + (long)d * PLANE + (n & (PLANE - 1));
  out_zq[pos] = c;
  float diff = ze[pos] - c;
  float s = diff * diff;
  #pragma unroll
  for (int m = 16; m >= 1; m >>= 1) s += __shfl_xor(s, m, 32);
  if ((threadIdx.x & 31) == 0) partial[threadIdx.x >> 5] = s;
  __syncthreads();
  if (threadIdx.x == 0) {
    float t = 0.f;
    #pragma unroll
    for (int w = 0; w < 8; ++w) t += partial[w];
    blocksums[n] = t;
  }
}

// --- Kernel 4: deterministic final reduction of the commitment loss --------
__global__ void vq_finalize(const float* __restrict__ blocksums, float* __restrict__ out_loss) {
  __shared__ float sm[256];
  float s = 0.f;
  for (int i = threadIdx.x; i < VQ_N; i += 256) s += blocksums[i];
  sm[threadIdx.x] = s;
  __syncthreads();
  for (int st = 128; st > 0; st >>= 1) {
    if (threadIdx.x < st) sm[threadIdx.x] += sm[threadIdx.x + st];
    __syncthreads();
  }
  if (threadIdx.x == 0) out_loss[0] = 0.25f * sm[0] / 2097152.0f;
}

extern "C" void kernel_launch(void* const* d_in, const int* in_sizes, int n_in,
                              void* d_out, int out_size, void* d_ws, size_t ws_size,
                              hipStream_t stream) {
  const float* ze = (const float*)d_in[0];   // [4,256,32,64]
  const float* cb = (const float*)d_in[1];   // [8192,256]

  float* out      = (float*)d_out;
  float* out_idx  = out;                         // 8192 (indices as float)
  float* out_zq   = out + VQ_N;                  // 2097152
  float* out_loss = out + VQ_N + 2097152;        // 1

  float* ws_csqr = (float*)d_ws;                 // 8192 f32
  int*   ws_idx  = (int*)((float*)d_ws + VQ_N);  // 8192 i32
  float* ws_bs   = (float*)d_ws + 2 * VQ_N;      // 8192 f32

  vq_csqr   <<<VQ_K / 8, 256, 0, stream>>>(cb, ws_csqr);
  vq_argmin <<<VQ_N / 64, 128, 0, stream>>>(ze, cb, ws_csqr, out_idx, ws_idx);
  vq_gather <<<VQ_N, 256, 0, stream>>>(ze, cb, ws_idx, out_zq, ws_bs);
  vq_finalize<<<1, 256, 0, stream>>>(ws_bs, out_loss);
}
